// RFFTConvTranspose_84542136254837
// MI455X (gfx1250) — matrix operand order repaired and checked by reading, not compiled, not run on hardware
//
#include <hip/hip_runtime.h>

// ---------------------------------------------------------------------------
// RFFT2 -> complex 3x3 conv (16ch->16ch, WMMA bf16-split fp32) -> IRFFT2
// MI455X / gfx1250, wave32.
//  - conv input slab stored pre-split as bf16 hi/lo (same 128B/position as f32)
//  - WMMA A fragments pre-baked once into a 20KB table by a 1-wave prep kernel
//  -> conv inner loop = pure b128 loads + v_wmma, no conversion VALU
// ---------------------------------------------------------------------------

typedef __attribute__((ext_vector_type(16))) __bf16 v16bf;
typedef __attribute__((ext_vector_type(8)))  float  v8f;

#define HH  512
#define WW  512
#define WF  257     // 512/2 + 1
#define NCH 16
#define NB  8

static __device__ __forceinline__ unsigned short f2bf(float f) {
  unsigned int u = __float_as_uint(f);
  u += 0x7fffu + ((u >> 16) & 1u);     // round-to-nearest-even
  return (unsigned short)(u >> 16);
}
static __device__ __forceinline__ float bf2f(unsigned short h) {
  return __uint_as_float(((unsigned int)h) << 16);
}

union bfu {
  v16bf v;
  unsigned short s[16];
  uint4 q[2];
};

static __device__ __forceinline__ v8f wmma_bf16(v16bf a, v16bf b, v8f c) {
  // (neg_a, A, neg_b, B, c_mod, C, reuse_a, reuse_b)
  return __builtin_amdgcn_wmma_f32_16x16x32_bf16(false, a, false, b,
                                                 (short)0, c, false, false);
}

// ---------------------------------------------------------------------------
// 512-point complex Stockham FFT in LDS (natural-order output, no bitrev).
// P threads per transform, 256/P butterflies each; element i lives at
// base[i*stride + off].  Ping-pongs b0<->b1; after 9 stages result is in b1.
// sgn = -1 forward, +1 inverse (unnormalized).
// ---------------------------------------------------------------------------
static __device__ __forceinline__ void fft512(float2* b0, float2* b1,
                                              int off, int stride,
                                              int tid, int P, float sgn)
{
  float2* x = b0; float2* y = b1;
  const int nloc = 256 / P;
  for (int stage = 0; stage < 9; ++stage) {
    const int ls = stage;
    const int s  = 1 << ls;
    const float base = sgn * 6.283185307179586f / (float)(512 >> stage);
    for (int r = 0; r < nloc; ++r) {
      const int t = tid + r * P;
      const int p = t >> ls;
      float2 a = x[t * stride + off];
      float2 b = x[(t + 256) * stride + off];
      float sn, cs;
      __sincosf(base * (float)p, &sn, &cs);
      float2 u, d, v;
      u.x = a.x + b.x; u.y = a.y + b.y;
      d.x = a.x - b.x; d.y = a.y - b.y;
      v.x = d.x * cs - d.y * sn;
      v.y = d.x * sn + d.y * cs;
      const int o = t + (p << ls);
      y[o * stride + off]       = u;
      y[(o + s) * stride + off] = v;
    }
    __syncthreads();
    float2* tmp = x; x = y; y = tmp;
  }
}

// ---------------------------------------------------------------------------
// K0: one-wave prep kernel.  Bakes WMMA A fragments (conv weights, bf16 hi/lo
// split, contiguous k = 16*laneHi + e per lane) into a 20KB table:
//   tab[((c*4 + frag)*32 + lane)*16 + e],  frag: 0=wr_hi 1=wr_lo 2=wi_hi 3=wi_lo
// ---------------------------------------------------------------------------
__global__ void wprep_k(const float* __restrict__ wre,
                        const float* __restrict__ wim,
                        unsigned short* __restrict__ tab)
{
  const int lane = threadIdx.x;             // 0..31
  const int laneHi = lane >> 4, ln = lane & 15;
  for (int c = 0; c < 5; ++c) {
    for (int e = 0; e < 16; ++e) {
      const int k = c * 32 + e + 16 * laneHi;
      float vr = 0.f, vi = 0.f;
      if (k < 144) {
        const int tap = k >> 4;             // kh*3+kw of equivalent conv
        const int ci  = k & 15;
        const int kh = tap / 3, kw = tap - kh * 3;
        // conv weight = transpose + spatial flip of ConvT weight (ci,co,kh,kw)
        const int idx = (ci * 16 + ln) * 9 + (2 - kh) * 3 + (2 - kw);
        vr = wre[idx];
        vi = wim[idx];
      }
      const unsigned short rh = f2bf(vr), rl = f2bf(vr - bf2f(rh));
      const unsigned short ih = f2bf(vi), il = f2bf(vi - bf2f(ih));
      tab[((c * 4 + 0) * 32 + lane) * 16 + e] = rh;
      tab[((c * 4 + 1) * 32 + lane) * 16 + e] = rl;
      tab[((c * 4 + 2) * 32 + lane) * 16 + e] = ih;
      tab[((c * 4 + 3) * 32 + lane) * 16 + e] = il;
    }
  }
}

// ---------------------------------------------------------------------------
// K1: per-row real->complex FFT.  x[row][0..511] -> slab0[row][0..256] cplx.
// One block per row (N*C*H = 65536 blocks).
// ---------------------------------------------------------------------------
__global__ __launch_bounds__(256) void rowfft_k(const float* __restrict__ x,
                                                float2* __restrict__ out)
{
  __shared__ float2 lds[2][512];
  const int row = blockIdx.x;
  const int t = threadIdx.x;
  const float* src = x + (size_t)row * WW;
  lds[0][t]       = make_float2(src[t], 0.f);
  lds[0][t + 256] = make_float2(src[t + 256], 0.f);
  __syncthreads();
  fft512(lds[0], lds[1], 0, 1, t, 256, -1.f);
  float2* dst = out + (size_t)row * WF;
  for (int w = t; w < WF; w += 256) dst[w] = lds[1][w];
}

// ---------------------------------------------------------------------------
// K2: column FFT (8 columns per block, 64KB LDS).  Output: channels-last,
// bf16 hi/lo pre-split, 128B per position (same bytes as f32 complex):
//   outS[pos*64 + {0,16,32,48} + ch] = {re_hi, re_lo, im_hi, im_lo}
// so the conv kernel's WMMA B fragments are direct 32B contiguous loads.
// ---------------------------------------------------------------------------
__global__ __launch_bounds__(256) void colfft_k(const float2* __restrict__ in,
                                                unsigned short* __restrict__ outS)
{
  __shared__ float2 lds[2][4096];           // [idx*8 + col]
  const int wg    = blockIdx.x % 33;
  const int plane = blockIdx.x / 33;        // n*16 + ch
  const int n = plane >> 4, ch = plane & 15;
  const int w0 = wg * 8;
  const int t = threadIdx.x;
  const float2* src = in + (size_t)plane * (HH * WF);
  #pragma unroll
  for (int k = 0; k < 16; ++k) {
    const int i = t + (k << 8);             // i == h*8 + c
    const int h = i >> 3, c = i & 7, w = w0 + c;
    lds[0][i] = (w < WF) ? src[(size_t)h * WF + w] : make_float2(0.f, 0.f);
  }
  __syncthreads();
  fft512(lds[0], lds[1], t & 7, 8, t >> 3, 32, -1.f);
  #pragma unroll
  for (int k = 0; k < 16; ++k) {
    const int i = t + (k << 8);
    const int h = i >> 3, c = i & 7, w = w0 + c;
    if (w < WF) {
      const float2 v = lds[1][i];
      const unsigned short rh = f2bf(v.x), rl = f2bf(v.x - bf2f(rh));
      const unsigned short ih = f2bf(v.y), il = f2bf(v.y - bf2f(ih));
      unsigned short* p = outS + (size_t)((n * HH + h) * WF + w) * 64;
      p[ch]      = rh;
      p[16 + ch] = rl;
      p[32 + ch] = ih;
      p[48 + ch] = il;
    }
  }
}

// ---------------------------------------------------------------------------
// K3: frequency-domain complex 3x3 conv, channels 16->16, via WMMA.
// GEMM per (n,h,16-wide w tile):  D[16co x 16w] = A[16 x 160] B[160 x 16],
// K = ci(16) x tap(9) padded to 160 = 5 K32 bf16 chunks.
// fp32-via-bf16 split: hi*hi + hi*lo + lo*hi (3 WMMAs / product term).
// 4 accumulators avoid A-negation:  Yr = RR - II,  Yi = IR + RI.
// A fragments from the prebaked table; B fragments are contiguous 32B loads
// from the pre-split slab.  One wave per tile, 8 waves per block.
// ---------------------------------------------------------------------------
__global__ __launch_bounds__(256) void freqconv_k(const unsigned short* __restrict__ fin,
                                                  float2* __restrict__ fout,
                                                  const unsigned short* __restrict__ wtab,
                                                  const float* __restrict__ bre,
                                                  const float* __restrict__ bim)
{
  __shared__ float lb[2][16];
  if (threadIdx.x < 16) {
    lb[0][threadIdx.x] = bre[threadIdx.x];
    lb[1][threadIdx.x] = bim[threadIdx.x];
  }
  __syncthreads();

  const int wave   = threadIdx.x >> 5;
  const int lane   = threadIdx.x & 31;
  const int laneHi = lane >> 4;
  const int ln     = lane & 15;

  int tile = blockIdx.x * 8 + wave;         // 8*512*17 = 69632 tiles, exact
  const int wg = tile % 17;
  tile /= 17;
  const int h = tile & 511;
  const int n = tile >> 9;

  v8f aRR = {0.f,0.f,0.f,0.f,0.f,0.f,0.f,0.f};
  v8f aII = aRR, aIR = aRR, aRI = aRR;

  #pragma unroll
  for (int c = 0; c < 5; ++c) {
    // ---- A fragments: 4 x 32B contiguous per lane from the baked table ----
    bfu Arh, Arl, Aih, Ail;
    {
      const uint4* ap = (const uint4*)(wtab + ((size_t)(c * 4) * 32 + lane) * 16);
      // frag f is at ap + f*32*16 ushorts = f*64 uint4s... index in uint4s:
      // ((c*4+f)*32+lane)*16 ushorts -> ap0 + f*(32*16/8) uint4
      Arh.q[0] = ap[0];  Arh.q[1] = ap[1];
      Arl.q[0] = ap[64]; Arl.q[1] = ap[65];     // +32 lanes * 16 ushorts = 64 uint4
      Aih.q[0] = ap[128]; Aih.q[1] = ap[129];
      Ail.q[0] = ap[192]; Ail.q[1] = ap[193];
    }
    // ---- B fragments: one tap x 16ci, 128B contiguous (pre-split) ----
    const int tap = 2 * c + laneHi;         // tap 9 (chunk4, hi half) = zero pad
    const int kh = tap / 3, kw = tap - kh * 3;
    const int hh  = h + kh - 1;
    const int wp  = wg * 16 + ln + kw - 1;
    const bool valid = (tap < 9) && (hh >= 0) && (hh < HH) && (wp >= 0) && (wp < WF);
    bfu Brh, Brl, Bih, Bil;
    if (valid) {
      const uint4* bp = (const uint4*)(fin + (size_t)((n * HH + hh) * WF + wp) * 64);
      Brh.q[0] = bp[0]; Brh.q[1] = bp[1];
      Brl.q[0] = bp[2]; Brl.q[1] = bp[3];
      Bih.q[0] = bp[4]; Bih.q[1] = bp[5];
      Bil.q[0] = bp[6]; Bil.q[1] = bp[7];
    } else {
      const uint4 z = make_uint4(0u, 0u, 0u, 0u);
      Brh.q[0] = z; Brh.q[1] = z;
      Brl.q[0] = z; Brl.q[1] = z;
      Bih.q[0] = z; Bih.q[1] = z;
      Bil.q[0] = z; Bil.q[1] = z;
    }
    // ---- 12 WMMAs per chunk ----
    aRR = wmma_bf16(Arh.v, Brh.v, aRR);
    aRR = wmma_bf16(Arh.v, Brl.v, aRR);
    aRR = wmma_bf16(Arl.v, Brh.v, aRR);
    aII = wmma_bf16(Aih.v, Bih.v, aII);
    aII = wmma_bf16(Aih.v, Bil.v, aII);
    aII = wmma_bf16(Ail.v, Bih.v, aII);
    aIR = wmma_bf16(Aih.v, Brh.v, aIR);
    aIR = wmma_bf16(Aih.v, Brl.v, aIR);
    aIR = wmma_bf16(Ail.v, Brh.v, aIR);
    aRI = wmma_bf16(Arh.v, Bih.v, aRI);
    aRI = wmma_bf16(Arh.v, Bil.v, aRI);
    aRI = wmma_bf16(Arl.v, Bih.v, aRI);
  }

  // C/D layout: VGPR v -> M = v + 8*laneHi (= co), N = lane&15 (= w in tile)
  const int w = wg * 16 + ln;
  if (w < WF) {
    float2* op = fout + ((size_t)((n * HH + h) * WF + w) << 4) + 8 * laneHi;
    #pragma unroll
    for (int v2 = 0; v2 < 8; ++v2) {
      const int co = v2 + 8 * laneHi;
      float2 o;
      o.x = aRR[v2] - aII[v2] + lb[0][co];
      o.y = aIR[v2] + aRI[v2] + lb[1][co];
      op[v2] = o;
    }
  }
}

// ---------------------------------------------------------------------------
// K4: inverse column FFT.  Gathers channels-last f32 conv output, writes
// plane-major complex rows with 1/512 scaling.
// ---------------------------------------------------------------------------
__global__ __launch_bounds__(256) void colifft_k(const float2* __restrict__ inCL,
                                                 float2* __restrict__ out)
{
  __shared__ float2 lds[2][4096];
  const int wg    = blockIdx.x % 33;
  const int plane = blockIdx.x / 33;
  const int n = plane >> 4, ch = plane & 15;
  const int w0 = wg * 8;
  const int t = threadIdx.x;
  #pragma unroll
  for (int k = 0; k < 16; ++k) {
    const int i = t + (k << 8);
    const int h = i >> 3, c = i & 7, w = w0 + c;
    lds[0][i] = (w < WF) ? inCL[((size_t)((n * HH + h) * WF + w) << 4) + ch]
                         : make_float2(0.f, 0.f);
  }
  __syncthreads();
  fft512(lds[0], lds[1], t & 7, 8, t >> 3, 32, +1.f);
  float2* dst = out + (size_t)plane * (HH * WF);
  const float sc = 1.f / 512.f;
  #pragma unroll
  for (int k = 0; k < 16; ++k) {
    const int i = t + (k << 8);
    const int h = i >> 3, c = i & 7, w = w0 + c;
    if (w < WF) {
      float2 v = lds[1][i];
      dst[(size_t)h * WF + w] = make_float2(v.x * sc, v.y * sc);
    }
  }
}

// ---------------------------------------------------------------------------
// K5: inverse real row FFT.  Rebuild full 512 spectrum from 257 bins via
// Hermitian symmetry, inverse FFT, keep real part * 1/512.
// ---------------------------------------------------------------------------
__global__ __launch_bounds__(256) void rowifft_k(const float2* __restrict__ in,
                                                 float* __restrict__ out)
{
  __shared__ float2 lds[2][512];
  const int row = blockIdx.x;
  const int t = threadIdx.x;
  const float2* src = in + (size_t)row * WF;
  for (int i = t; i < 512; i += 256) {
    float2 v;
    if (i < WF) { v = src[i]; }
    else        { v = src[512 - i]; v.y = -v.y; }
    lds[0][i] = v;
  }
  __syncthreads();
  fft512(lds[0], lds[1], 0, 1, t, 256, +1.f);
  float* dst = out + (size_t)row * WW;
  const float sc = 1.f / 512.f;
  dst[t]       = lds[1][t].x * sc;
  dst[t + 256] = lds[1][t + 256].x * sc;
}

// ---------------------------------------------------------------------------
extern "C" void kernel_launch(void* const* d_in, const int* in_sizes, int n_in,
                              void* d_out, int out_size, void* d_ws, size_t ws_size,
                              hipStream_t stream)
{
  (void)in_sizes; (void)n_in; (void)out_size; (void)ws_size;
  const float* x   = (const float*)d_in[0];   // [8,16,512,512]
  const float* wre = (const float*)d_in[1];   // [16,16,3,3] (ci,co,kh,kw)
  const float* wim = (const float*)d_in[2];
  const float* bre = (const float*)d_in[3];   // [16]
  const float* bim = (const float*)d_in[4];
  float* out = (float*)d_out;

  const size_t S = (size_t)NB * NCH * HH * WF;       // elems per slab
  char* ws = (char*)d_ws;
  float2*         slab0 = (float2*)ws;                          // f32 complex slab
  unsigned short* slab1s = (unsigned short*)(ws + S * 8);       // bf16-split slab
  float2*         slab1f = (float2*)(ws + S * 8);               // aliased f32 slab
  unsigned short* wtab   = (unsigned short*)(ws + 2 * S * 8);   // 20KB A-frag table

  wprep_k   <<<1, 32, 0, stream>>>(wre, wim, wtab);
  rowfft_k  <<<NB * NCH * HH, 256, 0, stream>>>(x, slab0);
  colfft_k  <<<NB * NCH * 33, 256, 0, stream>>>(slab0, slab1s);
  freqconv_k<<<(NB * HH * 17) / 8, 256, 0, stream>>>(slab1s, slab0,
                                                     wtab, bre, bim);
  colifft_k <<<NB * NCH * 33, 256, 0, stream>>>(slab0, slab1f);
  rowifft_k <<<NB * NCH * HH, 256, 0, stream>>>(slab1f, out);
}
